// GRUSubLayer_60833916780787
// MI455X (gfx1250) — matrix operand order repaired and checked by reading, not compiled, not run on hardware
//
#include <hip/hip_runtime.h>
#include <hip/hip_bf16.h>
#include <math.h>
#include <stdint.h>

typedef __attribute__((ext_vector_type(16))) __bf16 v16bf;
typedef __attribute__((ext_vector_type(8)))  __bf16 v8bf;
typedef __attribute__((ext_vector_type(8)))  float  v8f;

constexpr int kB  = 64;
constexpr int kT  = 512;
constexpr int kDI = 1024;
constexpr int kDH = 1024;
constexpr int kG3 = 3 * kDH;   // z|r|n gate columns packed side by side

// ---------------------------------------------------------------------------
// helpers
// ---------------------------------------------------------------------------
__device__ __forceinline__ float sigmoid_f(float x) {
  return 1.0f / (1.0f + __expf(-x));
}

// A fragment (16x32 bf16, M x K), CDNA5 16-bit A layout:
// lanes 0-15: M=lane, K = {0..7, 16..23}; lanes 16-31: M=lane-16, K = {8..15, 24..31}
__device__ __forceinline__ v16bf load_a_frag(const __bf16* __restrict__ A,
                                             int lda, int lane) {
  const int m  = lane & 15;
  const int kb = (lane >> 4) << 3;            // 0 or 8
  const __bf16* p = A + (size_t)m * lda + kb;
  v8bf lo = *(const v8bf*)(p);                // K = kb .. kb+7
  v8bf hi = *(const v8bf*)(p + 16);           // K = kb+16 .. kb+23
  v16bf a;
#pragma unroll
  for (int i = 0; i < 8; ++i) { a[i] = lo[i]; a[8 + i] = hi[i]; }
  return a;
}

__device__ __forceinline__ v16bf combine_frag(v8bf lo, v8bf hi) {
  v16bf b;
#pragma unroll
  for (int i = 0; i < 8; ++i) { b[i] = lo[i]; b[8 + i] = hi[i]; }
  return b;
}

__device__ __forceinline__ v8f wmma_bf16(v16bf a, v16bf b, v8f c) {
  // (neg_a, A, neg_b, B, c_mod, C, reuse_a, reuse_b)
  return __builtin_amdgcn_wmma_f32_16x16x32_bf16(false, a, false, b,
                                                 (short)0, c, false, false);
}

// LDS byte offset for async-to-LDS VDST operand: generic->LDS keeps the
// offset in addr[31:0] (ISA sec. 10.2 aperture mapping).
__device__ __forceinline__ unsigned lds_off(const void* p) {
  return (unsigned)(uintptr_t)p;
}

// One 16B global -> LDS async copy (ASYNCcnt-tracked).
__device__ __forceinline__ void async_copy_b128(unsigned dst_lds,
                                                const void* src) {
  asm volatile("global_load_async_to_lds_b128 %0, %1, off"
               :: "v"(dst_lds), "v"(src) : "memory");
}

// ---------------------------------------------------------------------------
// prep kernels
// ---------------------------------------------------------------------------
__global__ __launch_bounds__(256)
void cvt_bf16_kernel(const float* __restrict__ src, __bf16* __restrict__ dst,
                     size_t n) {
  size_t i = (size_t)blockIdx.x * blockDim.x + threadIdx.x;
  size_t stride = (size_t)gridDim.x * blockDim.x;
  for (; i < n; i += stride) dst[i] = (__bf16)src[i];
}

// Split each W (2048x1024) into x-part / h-part and pack z|r|n as 3072 columns.
__global__ __launch_bounds__(256)
void pack_w_kernel(const float* __restrict__ Wz, const float* __restrict__ Wr,
                   const float* __restrict__ Wn,
                   __bf16* __restrict__ Wxb, __bf16* __restrict__ Whb) {
  size_t i = (size_t)blockIdx.x * blockDim.x + threadIdx.x;  // over kDI*kDH
  if (i >= (size_t)kDI * kDH) return;
  const int k = (int)(i / kDH);
  const int n = (int)(i % kDH);
  const size_t sx = (size_t)k * kDH + n;            // rows [0, DI)
  const size_t sh = (size_t)(k + kDI) * kDH + n;    // rows [DI, DI+DH)
  const size_t dr = (size_t)k * kG3;
  Wxb[dr + n]            = (__bf16)Wz[sx];
  Wxb[dr + kDH + n]      = (__bf16)Wr[sx];
  Wxb[dr + 2 * kDH + n]  = (__bf16)Wn[sx];
  Whb[dr + n]            = (__bf16)Wz[sh];
  Whb[dr + kDH + n]      = (__bf16)Wr[sh];
  Whb[dr + 2 * kDH + n]  = (__bf16)Wn[sh];
}

__global__ __launch_bounds__(256)
void init_h_kernel(const float* __restrict__ H0, float* __restrict__ Hf,
                   __bf16* __restrict__ Hb, unsigned* __restrict__ counter) {
  size_t i = (size_t)blockIdx.x * blockDim.x + threadIdx.x;
  if (i == 0) *counter = 0u;   // reset grid-barrier counter every launch
  if (i < (size_t)kB * kDH) {
    float h = H0[i];
    Hf[i] = h;
    Hb[i] = (__bf16)h;
  }
}

__global__ __launch_bounds__(256)
void copy_f32_kernel(const float* __restrict__ src, float* __restrict__ dst,
                     size_t n) {
  size_t i = (size_t)blockIdx.x * blockDim.x + threadIdx.x;
  if (i < n) dst[i] = src[i];
}

// ---------------------------------------------------------------------------
// Precompute Gx[b*T+t, 0:3072] = x_t @ [Wz_x|Wr_x|Wn_x] + [bz|br|bn]
// Grid: (512, 48), block = 128 (4 waves). Wave computes a 16(M) x 64(N) tile.
// B tiles (shared by all 4 waves) are staged through LDS with double-buffered
// global_load_async_to_lds_b128 copies.
// ---------------------------------------------------------------------------
__global__ __launch_bounds__(128)
void gemm_x_kernel(const __bf16* __restrict__ Xb, const __bf16* __restrict__ Wxb,
                   const float* __restrict__ bz, const float* __restrict__ br,
                   const float* __restrict__ bn, float* __restrict__ Gx) {
  __shared__ v8bf sLo[2][4][32];   // [buf][j-subtile][k-row] 16B each (4 KB)
  __shared__ v8bf sHi[2][4][32];   // (4 KB)

  const int tid  = threadIdx.x;
  const int lane = tid & 31;
  const int wave = tid >> 5;
  const int m0 = (blockIdx.x * 4 + wave) * 16;   // 0 .. 32752
  const int n0 = blockIdx.y * 64;                // 0 .. 3008

  auto stage = [&](int buf, int k0) {
    // 4 j x 32 k x 2 halves = 256 x 16B transfers, 2 per thread
    for (int i = tid; i < 256; i += 128) {
      const int half = i >> 7;
      const int r = i & 127;
      const int j = r >> 5;
      const int k = r & 31;
      const __bf16* src = Wxb + (size_t)(k0 + k) * kG3 + n0 + j * 16 + half * 8;
      const unsigned dst =
          lds_off(half ? (const void*)&sHi[buf][j][k] : (const void*)&sLo[buf][j][k]);
      async_copy_b128(dst, src);
    }
  };

  v8f acc[4] = {};
  const __bf16* Arow = Xb + (size_t)m0 * kDI;

  stage(0, 0);
  for (int k0 = 0; k0 < kDI; k0 += 32) {
    const int buf = (k0 >> 5) & 1;
    if (k0 + 32 < kDI) {
      stage(buf ^ 1, k0 + 32);
      // 2 outstanding from the stage just issued; <=2 -> previous tile done
      asm volatile("s_wait_asynccnt 0x2" ::: "memory");
    } else {
      asm volatile("s_wait_asynccnt 0x0" ::: "memory");
    }
    __syncthreads();

    v16bf a = load_a_frag(Arow + k0, kDI, lane);
#pragma unroll
    for (int j = 0; j < 4; ++j) {
      v16bf b;
#pragma unroll
      for (int s = 0; s < 16; ++s) {
        const int kk = (s & 7) + ((lane >> 4) << 3) + ((s >> 3) << 4);
        b[s] = ((lane & 8) ? sHi[buf][j][kk] : sLo[buf][j][kk])[lane & 7];
      }
      acc[j] = wmma_bf16(a, b, acc[j]);
    }
    __syncthreads();   // WAR: don't let next stage overwrite a live buffer
  }

  const int mrow = (lane >> 4) << 3;   // 0 or 8
  const int nn   = lane & 15;
#pragma unroll
  for (int j = 0; j < 4; ++j) {
    const int col = n0 + 16 * j + nn;
    const float bias = (col < kDH)     ? bz[col]
                     : (col < 2 * kDH) ? br[col - kDH]
                                       : bn[col - 2 * kDH];
#pragma unroll
    for (int r = 0; r < 8; ++r) {
      const int m = m0 + mrow + r;
      Gx[(size_t)m * kG3 + col] = acc[j][r] + bias;
    }
  }
}

// ---------------------------------------------------------------------------
// Persistent recurrent scan. 64 blocks x 128 threads (4 waves).
// Block owns output columns [16*blockIdx.x, +16); wave w owns rows [16w, +16).
// The block's slice of Wh (3 gates x 1024 K x 16 cols = 96 KB bf16) is
// invariant across all 512 steps: stage it into LDS once via async copies,
// then every step's 3-gate GEMM reads B from LDS only.
// ---------------------------------------------------------------------------
__global__ __launch_bounds__(128)
void gru_scan_kernel(const __bf16* __restrict__ Whb, const float* __restrict__ Gx,
                     float* __restrict__ Hf0, float* __restrict__ Hf1,
                     __bf16* __restrict__ Hb0, __bf16* __restrict__ Hb1,
                     float* __restrict__ Out, unsigned* __restrict__ counter) {
  __shared__ v8bf sLo[3 * 1024];   // 48 KB: cols n0..n0+7 per (gate,k)
  __shared__ v8bf sHi[3 * 1024];   // 48 KB: cols n0+8..n0+15 per (gate,k)

  const int lane = threadIdx.x & 31;
  const int wave = threadIdx.x >> 5;
  const int m0 = wave * 16;
  const int n0 = blockIdx.x * 16;
  const unsigned nb = gridDim.x;
  const int mrow = (lane >> 4) << 3;
  const int nn   = lane & 15;

  // ---- one-time async stage of the whole B panel (6144 x 16B = 96 KB) ----
  for (int i = threadIdx.x; i < 6144; i += 128) {
    const int half = i & 1;
    const int gk = i >> 1;          // g*1024 + k
    const int k = gk & 1023;
    const int g = gk >> 10;
    const __bf16* src = Whb + (size_t)k * kG3 + n0 + (g << 10) + (half << 3);
    const unsigned dst =
        lds_off(half ? (const void*)&sHi[gk] : (const void*)&sLo[gk]);
    async_copy_b128(dst, src);
  }
  asm volatile("s_wait_asynccnt 0x0" ::: "memory");
  __syncthreads();

  for (int t = 0; t < kT; ++t) {
    const __bf16* Hb  = (t & 1) ? Hb1 : Hb0;
    const float*  Hf  = (t & 1) ? Hf1 : Hf0;
    float*        HfN = (t & 1) ? Hf0 : Hf1;
    __bf16*       HbN = (t & 1) ? Hb0 : Hb1;

    v8f az = {}, ar = {}, an = {};
    const __bf16* Arow = Hb + (size_t)m0 * kDH;
    for (int k0 = 0; k0 < kDH; k0 += 32) {
      if (k0 + 32 < kDH) __builtin_prefetch(Arow + k0 + 32, 0, 0);
      v16bf a = load_a_frag(Arow + k0, kDH, lane);
      const v8bf* sB = (lane & 8) ? sHi : sLo;
      v16bf b0, b1, b2;
#pragma unroll
      for (int s = 0; s < 16; ++s) {
        const int base = k0 + (s & 7) + ((lane >> 4) << 3) + ((s >> 3) << 4);
        b0[s] = sB[base][lane & 7];
        b1[s] = sB[1024 + base][lane & 7];
        b2[s] = sB[2048 + base][lane & 7];
      }
      az = wmma_bf16(a, b0, az);
      ar = wmma_bf16(a, b1, ar);
      an = wmma_bf16(a, b2, an);
    }

#pragma unroll
    for (int r = 0; r < 8; ++r) {
      const int m = m0 + mrow + r;          // batch row
      const int n = n0 + nn;                // hidden column
      const size_t g = ((size_t)m * kT + t) * kG3 + n;
      const float zx = Gx[g];
      const float rx = Gx[g + kDH];
      const float nx = Gx[g + 2 * kDH];
      const float Z  = sigmoid_f(zx + az[r]);
      const float R  = sigmoid_f(rx + ar[r]);
      const float Ht = tanhf(nx + R * an[r]);
      const float Hp = Hf[(size_t)m * kDH + n];
      const float Hn = (1.0f - Z) * Hp + Z * Ht;
      HfN[(size_t)m * kDH + n] = Hn;
      HbN[(size_t)m * kDH + n] = (__bf16)Hn;
      Out[((size_t)m * kT + t) * kDH + n] = Hn;
    }

    // ---- grid-wide barrier (monotonic counter, device scope) ----
    __syncthreads();
    if (threadIdx.x == 0) {
      __threadfence();
      atomicAdd(counter, 1u);
      const unsigned expected = (unsigned)(t + 1) * nb;
      while (__hip_atomic_load(counter, __ATOMIC_ACQUIRE,
                               __HIP_MEMORY_SCOPE_AGENT) < expected) {
        __builtin_amdgcn_s_sleep(2);
      }
    }
    __syncthreads();
  }
}

// ---------------------------------------------------------------------------
// launch
// ---------------------------------------------------------------------------
extern "C" void kernel_launch(void* const* d_in, const int* in_sizes, int n_in,
                              void* d_out, int out_size, void* d_ws,
                              size_t ws_size, hipStream_t stream) {
  const float* X  = (const float*)d_in[0];
  const float* H0 = (const float*)d_in[1];
  const float* Wz = (const float*)d_in[2];
  const float* bz = (const float*)d_in[3];
  const float* Wr = (const float*)d_in[4];
  const float* br = (const float*)d_in[5];
  const float* Wn = (const float*)d_in[6];
  const float* bn = (const float*)d_in[7];
  float* Out = (float*)d_out;

  char* ws = (char*)d_ws;
  size_t off = 0;
  auto alloc = [&](size_t bytes) -> void* {
    void* p = ws + off;
    off += (bytes + 255) & ~(size_t)255;
    return p;
  };
  __bf16* Xb  = (__bf16*)alloc((size_t)kB * kT * kDI * 2);   //  64 MB
  __bf16* Wxb = (__bf16*)alloc((size_t)kDI * kG3 * 2);       //   6 MB
  __bf16* Whb = (__bf16*)alloc((size_t)kDH * kG3 * 2);       //   6 MB
  float*  Gx  = (float*) alloc((size_t)kB * kT * kG3 * 4);   // 384 MB
  float*  Hf0 = (float*) alloc((size_t)kB * kDH * 4);
  float*  Hf1 = (float*) alloc((size_t)kB * kDH * 4);
  __bf16* Hb0 = (__bf16*)alloc((size_t)kB * kDH * 2);
  __bf16* Hb1 = (__bf16*)alloc((size_t)kB * kDH * 2);
  unsigned* counter = (unsigned*)alloc(256);

  cvt_bf16_kernel<<<4096, 256, 0, stream>>>(X, Xb, (size_t)kB * kT * kDI);
  pack_w_kernel<<<((size_t)kDI * kDH + 255) / 256, 256, 0, stream>>>(
      Wz, Wr, Wn, Wxb, Whb);
  init_h_kernel<<<(kB * kDH + 255) / 256, 256, 0, stream>>>(H0, Hf0, Hb0,
                                                            counter);
  gemm_x_kernel<<<dim3(512, 48), 128, 0, stream>>>(Xb, Wxb, bz, br, bn, Gx);
  gru_scan_kernel<<<64, 128, 0, stream>>>(Whb, Gx, Hf0, Hf1, Hb0, Hb1, Out,
                                          counter);
  // final hidden state lives in buffer 0 after 512 steps (t=511 writes Hf0)
  copy_f32_kernel<<<(kB * kDH + 255) / 256, 256, 0, stream>>>(
      Hf0, Out + (size_t)kB * kT * kDH, (size_t)kB * kDH);
}
